// Encoder_88940182766140
// MI455X (gfx1250) — matrix operand order repaired and checked by reading, not compiled, not run on hardware
//
#include <hip/hip_runtime.h>
#include <hip/hip_bf16.h>
#include <stdint.h>

// ---------------------------------------------------------------------------
// Point-Transformer encoder for MI455X (gfx1250, wave32, WMMA).
// GEMMs: v_wmma_f32_16x16x32_f16, 128x64 block tile, 8 waves x 4 acc tiles,
// double-buffered LDS staging via global_load_async_to_lds_b128 (inline asm)
// with s_wait_asynccnt pipelining. Weights pre-transposed to [N][K] f16 so
// both operands stream as contiguous 16B async chunks.
// ---------------------------------------------------------------------------

#define WB 4
#define WN 4096
#define WK 16
#define WDM 256

typedef __attribute__((ext_vector_type(16))) _Float16 v16h;
typedef __attribute__((ext_vector_type(8)))  _Float16 v8h;
typedef __attribute__((ext_vector_type(8)))  float    v8f;

#define HAVE_ASYNC 1

__device__ __forceinline__ void async_cp16(const void* g, void* l) {
#if HAVE_ASYNC
  // LDS aperture lives in addr[63:32]; addr[31:0] of a flat shared pointer is
  // the wave-relative LDS byte address (ISA 10.2) -> VDST operand.
  unsigned lds = (unsigned)(unsigned long long)l;
  unsigned long long ga = (unsigned long long)g;
  asm volatile("global_load_async_to_lds_b128 %0, %1, off"
               :: "v"(lds), "v"(ga) : "memory");
#else
  *(v8h*)l = *(const v8h*)g;
#endif
}

template <int N>
__device__ __forceinline__ void async_wait_le() {
#if HAVE_ASYNC
#if __has_builtin(__builtin_amdgcn_s_wait_asynccnt)
  __builtin_amdgcn_s_wait_asynccnt(N);
#else
  if (N == 0) asm volatile("s_wait_asynccnt 0" ::: "memory");
  else        asm volatile("s_wait_asynccnt 3" ::: "memory");
#endif
#endif
}

// ---------------------------------------------------------------------------
// WMMA GEMM:  C[M,Nd] = act(A[M,Kp]_f16 @ W[Kp,Nd]_f16 + bias) (+resid)
// A: [M][Kp] row-major f16 (M padded to 128).
// WT: [Nd64][Kp] f16 = W transposed, N padded to 64, pad rows zeroed.
// Block: 256 threads = 8 waves; tile 128(M) x 64(N); K-step 32; LDS ping-pong.
// ---------------------------------------------------------------------------
__global__ void __launch_bounds__(256)
k_gemm(const _Float16* __restrict__ A, const _Float16* __restrict__ WT,
       const float* __restrict__ bias, const float* __restrict__ resid,
       _Float16* __restrict__ outH, float* __restrict__ outF,
       int M, int Kp, int Nd, int relu)
{
  __shared__ alignas(16) _Float16 sA[2][128 * 32];  // [row][k], 32 halves/row
  __shared__ alignas(16) _Float16 sW[2][64 * 32];   // [n][k],   32 halves/row

  const int tid  = threadIdx.x;
  const int wave = tid >> 5;
  const int lane = tid & 31;
  const int rowBase = blockIdx.x * 128;
  const int colBase = blockIdx.y * 64;
  const int fm = lane & 15;   // A row in wave tile / B column in n-tile
  const int kh = lane >> 4;   // K half selector
  const int nk = Kp >> 5;

  v8f acc[4] = {};

  // Stage one 32-wide K slab of A (8KB) + WT (4KB): 3 async 16B chunks/thread.
  auto stage = [&](int buf, int ks) {
#pragma unroll
    for (int it = 0; it < 2; ++it) {
      int chunk = it * 256 + tid;               // 0..511
      int r = chunk >> 2, c = chunk & 3;
      async_cp16(A + (size_t)(rowBase + r) * Kp + ks + c * 8,
                 &sA[buf][r * 32 + c * 8]);
    }
    {
      int r = tid >> 2, c = tid & 3;            // 0..63 x 0..3
      async_cp16(WT + (size_t)(colBase + r) * Kp + ks + c * 8,
                 &sW[buf][r * 32 + c * 8]);
    }
  };

  stage(0, 0);
  for (int ksi = 0; ksi < nk; ++ksi) {
    const int buf = ksi & 1;
    if (ksi + 1 < nk) { stage(buf ^ 1, (ksi + 1) << 5); async_wait_le<3>(); }
    else              { async_wait_le<0>(); }
    __syncthreads();

    // A fragment (16-bit A 16x32 layout)
    v16h af;
    {
      const v8h* p0 = (const v8h*)(&sA[buf][(wave * 16 + fm) * 32 + kh * 8]);
      const v8h* p1 = (const v8h*)(&sA[buf][(wave * 16 + fm) * 32 + 16 + kh * 8]);
      v8h lo = *p0, hi = *p1;
#pragma unroll
      for (int i = 0; i < 8; ++i) { af[i] = lo[i]; af[8 + i] = hi[i]; }
    }
    // 4 N-tiles share the A fragment
#pragma unroll
    for (int t = 0; t < 4; ++t) {
      v16h bf;
      const v8h* q0 = (const v8h*)(&sW[buf][(t * 16 + fm) * 32 + kh * 8]);
      const v8h* q1 = (const v8h*)(&sW[buf][(t * 16 + fm) * 32 + 16 + kh * 8]);
      v8h lo = *q0, hi = *q1;
#pragma unroll
      for (int i = 0; i < 8; ++i) { bf[i] = lo[i]; bf[8 + i] = hi[i]; }
      acc[t] = __builtin_amdgcn_wmma_f32_16x16x32_f16(false, af, false, bf,
                                                      (short)0, acc[t], false, false);
    }
    __syncthreads();
  }

  // epilogue: C 16x16 layout: lane(n=l&15, mh=l>>4), VGPR i -> M = i + 8*mh
  const int n  = lane & 15;
  const int mh = lane >> 4;
#pragma unroll
  for (int t = 0; t < 4; ++t) {
    if (colBase + t * 16 < Nd) {
      const int col = colBase + t * 16 + n;
      const float bv = bias ? bias[col] : 0.f;
#pragma unroll
      for (int i = 0; i < 8; ++i) {
        int row = rowBase + wave * 16 + mh * 8 + i;
        if (row >= M) continue;
        float v = acc[t][i] + bv;
        if (relu) v = fmaxf(v, 0.f);
        if (resid) v += resid[(size_t)row * Nd + col];
        if (outF) outF[(size_t)row * Nd + col] = v;
        if (outH) outH[(size_t)row * Nd + col] = (_Float16)v;
      }
    }
  }
}

// ---------------------------------------------------------------------------
// Small kernels
// ---------------------------------------------------------------------------
// pack W[Kd][Nd] f32 -> WT[Nd64][Kp] f16 (transposed, zero padded)
__global__ void k_wpack(const float* __restrict__ w, _Float16* __restrict__ o,
                        int Kd, int Nd, int Kp, int Nd64)
{
  int i = blockIdx.x * 256 + threadIdx.x;
  int tot = Nd64 * Kp;
  if (i >= tot) return;
  int n = i / Kp, k = i - n * Kp;
  o[i] = (n < Nd && k < Kd) ? (_Float16)w[(size_t)k * Nd + n] : (_Float16)0.f;
}

__global__ void k_xyz(const float* __restrict__ x, float* __restrict__ xyz, int M)
{
  int e = blockIdx.x * 256 + threadIdx.x;
  if (e >= M * 3) return;
  int p = e / 3, d = e - p * 3;
  xyz[e] = x[(size_t)p * 4 + d];
}

// stem: relu(x@w1+b1)@w2+b2   (4 -> 32 -> 32), writes f32 and f16 copies
__global__ void k_stem(const float* __restrict__ x,
                       const float* __restrict__ w1, const float* __restrict__ b1,
                       const float* __restrict__ w2, const float* __restrict__ b2,
                       _Float16* __restrict__ oh, float* __restrict__ of, int M)
{
  int p = blockIdx.x * 256 + threadIdx.x;
  if (p >= M) return;
  float in[4];
#pragma unroll
  for (int d = 0; d < 4; ++d) in[d] = x[(size_t)p * 4 + d];
  float h1[32];
#pragma unroll
  for (int j = 0; j < 32; ++j) {
    float s = b1[j];
#pragma unroll
    for (int d = 0; d < 4; ++d) s += in[d] * w1[d * 32 + j];
    h1[j] = fmaxf(s, 0.f);
  }
  for (int j = 0; j < 32; ++j) {
    float s = b2[j];
#pragma unroll
    for (int i = 0; i < 32; ++i) s += h1[i] * w2[i * 32 + j];
    of[(size_t)p * 32 + j] = s;
    oh[(size_t)p * 32 + j] = (_Float16)s;
  }
}

// KNN: per-query top-16 nearest refs, LDS-tiled candidate streaming.
__global__ void __launch_bounds__(256)
k_knn(const float* __restrict__ q, const float* __restrict__ r,
      int* __restrict__ idx, int Nq, int Nr)
{
  __shared__ float st[256 * 3];
  const int b  = blockIdx.y;
  const int qi = blockIdx.x * 256 + threadIdx.x;
  const bool act = (qi < Nq);
  float qx = 0, qy = 0, qz = 0;
  if (act) {
    qx = q[((size_t)b * Nq + qi) * 3 + 0];
    qy = q[((size_t)b * Nq + qi) * 3 + 1];
    qz = q[((size_t)b * Nq + qi) * 3 + 2];
  }
  float best[16]; int bi[16];
#pragma unroll
  for (int t = 0; t < 16; ++t) { best[t] = 3.4e38f; bi[t] = 0; }
  for (int base = 0; base < Nr; base += 256) {
    int nt = min(256, Nr - base);
    __syncthreads();
    if ((int)threadIdx.x < nt) {
      st[threadIdx.x * 3 + 0] = r[((size_t)b * Nr + base + threadIdx.x) * 3 + 0];
      st[threadIdx.x * 3 + 1] = r[((size_t)b * Nr + base + threadIdx.x) * 3 + 1];
      st[threadIdx.x * 3 + 2] = r[((size_t)b * Nr + base + threadIdx.x) * 3 + 2];
    }
    __syncthreads();
    if (act) {
      for (int j = 0; j < nt; ++j) {
        float dx = qx - st[j * 3 + 0];
        float dy = qy - st[j * 3 + 1];
        float dz = qz - st[j * 3 + 2];
        float d = dx * dx + dy * dy + dz * dz;
        if (d < best[15]) {
          int t = 15;
          while (t > 0 && best[t - 1] > d) { best[t] = best[t - 1]; bi[t] = bi[t - 1]; --t; }
          best[t] = d; bi[t] = base + j;
        }
      }
    }
  }
  if (act)
#pragma unroll
    for (int t = 0; t < 16; ++t) idx[((size_t)b * Nq + qi) * 16 + t] = bi[t];
}

// Farthest point sampling: one block per batch, LDS argmax reduction.
__global__ void __launch_bounds__(256)
k_fps(const float* __restrict__ xyz, int* __restrict__ fidx,
      float* __restrict__ dist, int Nn, int npoint)
{
  __shared__ float sv[256];
  __shared__ int   si[256];
  __shared__ int   sFar;
  const int b = blockIdx.x;
  const float* P = xyz + (size_t)b * Nn * 3;
  float* D = dist + (size_t)b * Nn;
  if (threadIdx.x == 0) sFar = 0;
  for (int i = threadIdx.x; i < Nn; i += 256) D[i] = 1e10f;
  __syncthreads();
  for (int s = 0; s < npoint; ++s) {
    int far = sFar;
    if (threadIdx.x == 0) fidx[(size_t)b * npoint + s] = far;
    float cx = P[far * 3 + 0], cy = P[far * 3 + 1], cz = P[far * 3 + 2];
    float bv = -1.f; int bj = 0;
    for (int i = threadIdx.x; i < Nn; i += 256) {
      float dx = P[i * 3 + 0] - cx, dy = P[i * 3 + 1] - cy, dz = P[i * 3 + 2] - cz;
      float nd = fminf(D[i], dx * dx + dy * dy + dz * dz);
      D[i] = nd;
      if (nd > bv) { bv = nd; bj = i; }
    }
    sv[threadIdx.x] = bv; si[threadIdx.x] = bj;
    __syncthreads();
    for (int off = 128; off > 0; off >>= 1) {
      if ((int)threadIdx.x < off && sv[threadIdx.x + off] > sv[threadIdx.x]) {
        sv[threadIdx.x] = sv[threadIdx.x + off];
        si[threadIdx.x] = si[threadIdx.x + off];
      }
      __syncthreads();
    }
    if (threadIdx.x == 0) sFar = si[0];
    __syncthreads();
  }
}

// vector gather (8 halves per thread): out[p][c8] = src[b*Nkv+idx[p]][c8]
__global__ void k_gather_h8(const _Float16* __restrict__ src, const int* __restrict__ idx,
                            _Float16* __restrict__ out, int NqK, int Nkv, int C8, int Ptot)
{
  int e = blockIdx.x * 256 + threadIdx.x;
  if (e >= Ptot * C8) return;
  int p = e / C8, c8 = e - p * C8;
  int b = p / NqK;
  const v8h* s = (const v8h*)(src + ((size_t)b * Nkv + idx[p]) * (size_t)(C8 * 8)) + c8;
  ((v8h*)out)[e] = *s;
}

__global__ void k_gather_f(const float* __restrict__ src, const int* __restrict__ idx,
                           float* __restrict__ out, int NqK, int Nkv, int C, int Ptot)
{
  int e = blockIdx.x * 256 + threadIdx.x;
  if (e >= Ptot * C) return;
  int p = e / C, c = e - p * C;
  int b = p / NqK;
  out[e] = src[((size_t)b * Nkv + idx[p]) * C + c];
}

// posin = relu(rel @ delta_w1 + delta_b1), rel = xyz_q - xyz_kv[idx]
__global__ void k_relpos(const float* __restrict__ xq, const float* __restrict__ xkv,
                         const int* __restrict__ idx,
                         const float* __restrict__ dw1, const float* __restrict__ db1,
                         _Float16* __restrict__ posin, int Nq, int Nkv, int Ptot)
{
  int e = blockIdx.x * 256 + threadIdx.x;
  if (e >= Ptot * 32) return;
  int p = e >> 5, g = (e & 31) * 8;
  int bn = p >> 4;              // K == 16
  int b  = bn / Nq;
  int r  = b * Nkv + idx[p];
  float rx = xq[(size_t)bn * 3 + 0] - xkv[(size_t)r * 3 + 0];
  float ry = xq[(size_t)bn * 3 + 1] - xkv[(size_t)r * 3 + 1];
  float rz = xq[(size_t)bn * 3 + 2] - xkv[(size_t)r * 3 + 2];
#pragma unroll
  for (int j = 0; j < 8; ++j) {
    int c = g + j;
    float s = db1[c] + rx * dw1[c] + ry * dw1[256 + c] + rz * dw1[512 + c];
    posin[(size_t)p * 256 + c] = (_Float16)fmaxf(s, 0.f);
  }
}

// h = q - k + pos ; vp = v + pos   (8 halves per thread)
__global__ void k_hvp8(const _Float16* __restrict__ q, const _Float16* __restrict__ kk,
                       const _Float16* __restrict__ v, const _Float16* __restrict__ pos,
                       _Float16* __restrict__ h, _Float16* __restrict__ vp, int Ptot)
{
  int e = blockIdx.x * 256 + threadIdx.x;
  if (e >= Ptot * 32) return;           // 32 x v8h per pair row (256 halves)
  int p = e >> 5;
  int bn = p >> 4;
  v8h qv = ((const v8h*)(q + (size_t)bn * 256))[e & 31];
  v8h kv = ((const v8h*)kk)[e];
  v8h vv = ((const v8h*)v)[e];
  v8h pv = ((const v8h*)pos)[e];
  v8h hh, w8;
#pragma unroll
  for (int i = 0; i < 8; ++i) {
    float pf = (float)pv[i];
    hh[i] = (_Float16)((float)qv[i] - (float)kv[i] + pf);
    w8[i] = (_Float16)((float)vv[i] + pf);
  }
  ((v8h*)h)[e]  = hh;
  ((v8h*)vp)[e] = w8;
}

// softmax over the 16 neighbors (axis=-2), scaled by 1/sqrt(256), then
// res[bn][c] = sum_k attn[k] * vp[k]
__global__ void k_smax(const float* __restrict__ logits, const _Float16* __restrict__ vp,
                       _Float16* __restrict__ resH, int Mq)
{
  int e = blockIdx.x * 256 + threadIdx.x;
  if (e >= Mq * 256) return;
  int bn = e >> 8, c = e & 255;
  const float* L = logits + ((size_t)bn * 16) * 256 + c;
  float l[16], mx = -3.4e38f;
#pragma unroll
  for (int k = 0; k < 16; ++k) { l[k] = L[(size_t)k * 256] * (1.f / 16.f); mx = fmaxf(mx, l[k]); }
  float s = 0.f;
#pragma unroll
  for (int k = 0; k < 16; ++k) { l[k] = __expf(l[k] - mx); s += l[k]; }
  float inv = 1.f / s;
  const _Float16* V = vp + ((size_t)bn * 16) * 256 + c;
  float acc = 0.f;
#pragma unroll
  for (int k = 0; k < 16; ++k) acc += l[k] * inv * (float)V[(size_t)k * 256];
  resH[e] = (_Float16)acc;
}

// transition_down: g = concat(xyz[idx]-new_xyz, pts[idx]) padded to Kp (f16)
__global__ void k_concat(const float* __restrict__ newxyz, const float* __restrict__ xyz,
                         const _Float16* __restrict__ pts, const int* __restrict__ idx,
                         _Float16* __restrict__ g, int Nq, int Nkv, int cin, int Kp, int Ptot)
{
  int e = blockIdx.x * 256 + threadIdx.x;
  if (e >= Ptot * Kp) return;
  int p = e / Kp, c = e - p * Kp;
  int bn = p >> 4;
  int b  = bn / Nq;
  int r  = b * Nkv + idx[p];
  _Float16 v;
  if (c < 3)            v = (_Float16)(xyz[(size_t)r * 3 + c] - newxyz[(size_t)bn * 3 + c]);
  else if (c < 3 + cin) v = pts[(size_t)r * cin + (c - 3)];
  else                  v = (_Float16)0.f;
  g[e] = v;
}

// max over the K=16 neighbors
__global__ void k_maxk(const _Float16* __restrict__ h, float* __restrict__ of,
                       _Float16* __restrict__ oh, int C, int Mq)
{
  int e = blockIdx.x * 256 + threadIdx.x;
  if (e >= Mq * C) return;
  int bn = e / C, c = e - bn * C;
  float m = -3.4e38f;
#pragma unroll
  for (int k = 0; k < 16; ++k) m = fmaxf(m, (float)h[((size_t)bn * 16 + k) * C + c]);
  of[e] = m;
  oh[e] = (_Float16)m;
}

// ---------------------------------------------------------------------------
// Host orchestration
// ---------------------------------------------------------------------------
static inline int pad32(int x) { return (x + 31) & ~31; }
static inline int pad64(int x) { return (x + 63) & ~63; }

struct Bump {
  char* base; size_t off;
  void* alloc(size_t b) { void* p = base + off; off = (off + b + 255) & ~(size_t)255; return p; }
};

struct SaW {
  const float *db1, *db2, *dw1, *dw2, *fc1_b, *fc1_w, *fc2_b, *fc2_w,
              *gb1, *gb2, *gw1, *gw2, *wk, *wq, *wv;
};
static SaW getSa(void* const* d, int b) {
  SaW s;
  s.db1   = (const float*)d[b + 0];  s.db2   = (const float*)d[b + 1];
  s.dw1   = (const float*)d[b + 2];  s.dw2   = (const float*)d[b + 3];
  s.fc1_b = (const float*)d[b + 4];  s.fc1_w = (const float*)d[b + 5];
  s.fc2_b = (const float*)d[b + 6];  s.fc2_w = (const float*)d[b + 7];
  s.gb1   = (const float*)d[b + 8];  s.gb2   = (const float*)d[b + 9];
  s.gw1   = (const float*)d[b + 10]; s.gw2   = (const float*)d[b + 11];
  s.wk    = (const float*)d[b + 12]; s.wq    = (const float*)d[b + 13];
  s.wv    = (const float*)d[b + 14];
  return s;
}

struct SaP { const _Float16 *fc1_w, *wq, *wk, *wv, *dw2, *gw1, *gw2, *fc2_w; };

static const _Float16* packW(const float* w, int Kd, int Nd, Bump& bp, hipStream_t st) {
  int Kp = pad32(Kd), Nd64 = pad64(Nd);
  _Float16* o = (_Float16*)bp.alloc((size_t)Kp * Nd64 * sizeof(_Float16));
  int tot = Kp * Nd64;
  k_wpack<<<(tot + 255) / 256, 256, 0, st>>>(w, o, Kd, Nd, Kp, Nd64);
  return o;
}

static SaP packSa(const SaW& w, int dp, Bump& bp, hipStream_t st) {
  SaP p;
  p.fc1_w = packW(w.fc1_w, dp, 256, bp, st);
  p.wq    = packW(w.wq, 256, 256, bp, st);
  p.wk    = packW(w.wk, 256, 256, bp, st);
  p.wv    = packW(w.wv, 256, 256, bp, st);
  p.dw2   = packW(w.dw2, 256, 256, bp, st);
  p.gw1   = packW(w.gw1, 256, 256, bp, st);
  p.gw2   = packW(w.gw2, 256, 256, bp, st);
  p.fc2_w = packW(w.fc2_w, 256, dp, bp, st);
  return p;
}

struct Pool {
  int* idx; int* fidx; float* fdist;
  _Float16 *xqH, *xkH, *qH, *kfH, *vfH, *resH;
  _Float16 *kkH, *vH, *posH, *hH, *vpH, *tH;
  float* logitsF;
};

static void gemm(hipStream_t st, const _Float16* A, const _Float16* WT, const float* bias,
                 const float* resid, _Float16* oh, float* of, int M, int Kp, int Nd, int relu) {
  dim3 g((M + 127) / 128, (Nd + 63) / 64);
  k_gemm<<<g, 256, 0, st>>>(A, WT, bias, resid, oh, of, M, Kp, Nd, relu);
}

static void attn(hipStream_t st, Pool& pl, const SaW& w, const SaP& wp,
                 const float* xyzQ, const float* xyzKV,
                 _Float16* fqH, float* fqF, const _Float16* fkvH,
                 float* outF, _Float16* outH, int Nq, int Nkv, int dp)
{
  const int Mq = WB * Nq, Mkv = WB * Nkv, P = Mq * WK;
  k_knn<<<dim3(Nq / 256, WB), 256, 0, st>>>(xyzQ, xyzKV, pl.idx, Nq, Nkv);
  gemm(st, fqH, wp.fc1_w, w.fc1_b, nullptr, pl.xqH, nullptr, Mq, pad32(dp), 256, 0);
  const _Float16* xkH = pl.xqH;
  if (fkvH != fqH) {
    gemm(st, fkvH, wp.fc1_w, w.fc1_b, nullptr, pl.xkH, nullptr, Mkv, pad32(dp), 256, 0);
    xkH = pl.xkH;
  }
  gemm(st, pl.xqH, wp.wq, nullptr, nullptr, pl.qH, nullptr, Mq, 256, 256, 0);
  gemm(st, xkH, wp.wk, nullptr, nullptr, pl.kfH, nullptr, Mkv, 256, 256, 0);
  gemm(st, xkH, wp.wv, nullptr, nullptr, pl.vfH, nullptr, Mkv, 256, 256, 0);
  k_gather_h8<<<(P * 32 + 255) / 256, 256, 0, st>>>(pl.kfH, pl.idx, pl.kkH, Nq * WK, Nkv, 32, P);
  k_gather_h8<<<(P * 32 + 255) / 256, 256, 0, st>>>(pl.vfH, pl.idx, pl.vH, Nq * WK, Nkv, 32, P);
  k_relpos<<<(P * 32 + 255) / 256, 256, 0, st>>>(xyzQ, xyzKV, pl.idx, w.dw1, w.db1, pl.tH, Nq, Nkv, P);
  gemm(st, pl.tH, wp.dw2, w.db2, nullptr, pl.posH, nullptr, P, 256, 256, 0);
  k_hvp8<<<(P * 32 + 255) / 256, 256, 0, st>>>(pl.qH, pl.kkH, pl.vH, pl.posH, pl.hH, pl.vpH, P);
  gemm(st, pl.hH, wp.gw1, w.gb1, nullptr, pl.tH, nullptr, P, 256, 256, 1);
  gemm(st, pl.tH, wp.gw2, w.gb2, nullptr, nullptr, pl.logitsF, P, 256, 256, 0);
  k_smax<<<(Mq * 256 + 255) / 256, 256, 0, st>>>(pl.logitsF, pl.vpH, pl.resH, Mq);
  gemm(st, pl.resH, wp.fc2_w, w.fc2_b, fqF, outH, outF, Mq, 256, dp, 0);
}

static void tdown(hipStream_t st, Pool& pl,
                  const float* b1, const float* b2,
                  const _Float16* w1p, const _Float16* w2p,
                  const float* xyzIn, const _Float16* ptsInH,
                  float* xyzOut, float* ptsOutF, _Float16* ptsOutH,
                  int Nin, int np, int cin, int ch)
{
  const int P = WB * np * WK;
  const int Kp = pad32(3 + cin);
  k_fps<<<WB, 256, 0, st>>>(xyzIn, pl.fidx, pl.fdist, Nin, np);
  k_gather_f<<<(WB * np * 3 + 255) / 256, 256, 0, st>>>(xyzIn, pl.fidx, xyzOut, np, Nin, 3, WB * np);
  k_knn<<<dim3(np / 256, WB), 256, 0, st>>>(xyzOut, xyzIn, pl.idx, np, Nin);
  k_concat<<<(P * Kp + 255) / 256, 256, 0, st>>>(xyzOut, xyzIn, ptsInH, pl.idx, pl.kkH, np, Nin, cin, Kp, P);
  gemm(st, pl.kkH, w1p, b1, nullptr, pl.vH, nullptr, P, Kp, ch, 1);
  gemm(st, pl.vH, w2p, b2, nullptr, pl.posH, nullptr, P, ch, ch, 1);
  k_maxk<<<(WB * np * ch + 255) / 256, 256, 0, st>>>(pl.posH, ptsOutF, ptsOutH, ch, WB * np);
}

extern "C" void kernel_launch(void* const* d_in, const int* in_sizes, int n_in,
                              void* d_out, int out_size, void* d_ws, size_t ws_size,
                              hipStream_t stream)
{
  (void)in_sizes; (void)n_in; (void)out_size; (void)ws_size;

  // --- input mapping: x1, x2, then params as JAX tree leaves (sorted keys):
  // cross(15), fc1_b1, fc1_b2, fc1_w1, fc1_w2, t1(15), td[0](b1,b2,w1,w2),
  // td[1](...), tf[0](15), tf[1](15)
  const float* x1 = (const float*)d_in[0];
  const float* x2 = (const float*)d_in[1];
  int pb = 2;
  SaW crossW = getSa(d_in, pb); pb += 15;
  const float* fc1_b1 = (const float*)d_in[pb++];
  const float* fc1_b2 = (const float*)d_in[pb++];
  const float* fc1_w1 = (const float*)d_in[pb++];
  const float* fc1_w2 = (const float*)d_in[pb++];
  SaW t1W = getSa(d_in, pb); pb += 15;
  const float *td_b1[2], *td_b2[2], *td_w1[2], *td_w2[2];
  for (int i = 0; i < 2; ++i) {
    td_b1[i] = (const float*)d_in[pb++];
    td_b2[i] = (const float*)d_in[pb++];
    td_w1[i] = (const float*)d_in[pb++];
    td_w2[i] = (const float*)d_in[pb++];
  }
  SaW tfW[2];
  tfW[0] = getSa(d_in, pb); pb += 15;
  tfW[1] = getSa(d_in, pb); pb += 15;

  Bump bp{(char*)d_ws, 0};

  // --- pack all GEMM weights to transposed f16 ---
  SaP t1P    = packSa(t1W, 32, bp, stream);
  SaP tfP[2] = { packSa(tfW[0], 64, bp, stream), packSa(tfW[1], 128, bp, stream) };
  SaP crossP = packSa(crossW, 128, bp, stream);
  const _Float16* tdw1p[2] = { packW(td_w1[0], 35, 64, bp, stream),
                               packW(td_w1[1], 67, 128, bp, stream) };
  const _Float16* tdw2p[2] = { packW(td_w2[0], 64, 64, bp, stream),
                               packW(td_w2[1], 128, 128, bp, stream) };

  // --- scratch pool (sized for level 0) ---
  const size_t Mmax = (size_t)WB * WN;      // 16384 rows
  const size_t Pmax = Mmax * WK;            // 262144 rows
  Pool pl;
  pl.idx   = (int*)bp.alloc(Pmax * sizeof(int));
  pl.fidx  = (int*)bp.alloc((size_t)WB * 1024 * sizeof(int));
  pl.fdist = (float*)bp.alloc((size_t)WB * WN * sizeof(float));
  pl.xqH  = (_Float16*)bp.alloc(Mmax * 256 * 2);
  pl.xkH  = (_Float16*)bp.alloc(Mmax * 256 * 2);
  pl.qH   = (_Float16*)bp.alloc(Mmax * 256 * 2);
  pl.kfH  = (_Float16*)bp.alloc(Mmax * 256 * 2);
  pl.vfH  = (_Float16*)bp.alloc(Mmax * 256 * 2);
  pl.resH = (_Float16*)bp.alloc(Mmax * 256 * 2);
  pl.kkH  = (_Float16*)bp.alloc(Pmax * 256 * 2);
  pl.vH   = (_Float16*)bp.alloc(Pmax * 256 * 2);
  pl.posH = (_Float16*)bp.alloc(Pmax * 256 * 2);
  pl.hH   = (_Float16*)bp.alloc(Pmax * 256 * 2);
  pl.vpH  = (_Float16*)bp.alloc(Pmax * 256 * 2);
  pl.tH   = (_Float16*)bp.alloc(Pmax * 256 * 2);
  pl.logitsF = (float*)bp.alloc(Pmax * 256 * 4);

  // --- per-level feature/coordinate buffers ---
  const int Nlev[3] = { WN, WN / 4, WN / 16 };
  const int dlev[3] = { 32, 64, 128 };
  float* xyzB[2][3]; float* ptsF[2][3]; _Float16* ptsH[2][3];
  for (int s = 0; s < 2; ++s)
    for (int l = 0; l < 3; ++l) {
      size_t Ml = (size_t)WB * Nlev[l];
      xyzB[s][l] = (float*)bp.alloc(Ml * 3 * sizeof(float));
      ptsF[s][l] = (float*)bp.alloc(Ml * dlev[l] * sizeof(float));
      ptsH[s][l] = (_Float16*)bp.alloc(Ml * dlev[l] * sizeof(_Float16));
    }
  float* crossF = (float*)bp.alloc((size_t)WB * 256 * 128 * sizeof(float));

  // --- pipeline ---
  const float* xin[2] = { x1, x2 };
  for (int s = 0; s < 2; ++s) {
    const int M0 = WB * WN;
    k_xyz<<<(M0 * 3 + 255) / 256, 256, 0, stream>>>(xin[s], xyzB[s][0], M0);
    k_stem<<<(M0 + 255) / 256, 256, 0, stream>>>(xin[s], fc1_w1, fc1_b1, fc1_w2, fc1_b2,
                                                 ptsH[s][0], ptsF[s][0], M0);
    // t1 self-attention (in place: residual/out are elementwise-aligned)
    attn(stream, pl, t1W, t1P, xyzB[s][0], xyzB[s][0], ptsH[s][0], ptsF[s][0],
         ptsH[s][0], ptsF[s][0], ptsH[s][0], WN, WN, 32);
    for (int i = 0; i < 2; ++i) {
      tdown(stream, pl, td_b1[i], td_b2[i], tdw1p[i], tdw2p[i],
            xyzB[s][i], ptsH[s][i], xyzB[s][i + 1], ptsF[s][i + 1], ptsH[s][i + 1],
            Nlev[i], Nlev[i + 1], dlev[i], dlev[i + 1]);
      attn(stream, pl, tfW[i], tfP[i], xyzB[s][i + 1], xyzB[s][i + 1],
           ptsH[s][i + 1], ptsF[s][i + 1], ptsH[s][i + 1],
           ptsF[s][i + 1], ptsH[s][i + 1], Nlev[i + 1], Nlev[i + 1], dlev[i + 1]);
    }
  }
  // cross attention: q-side = cloud 1, kv-side = cloud 2
  attn(stream, pl, crossW, crossP, xyzB[0][2], xyzB[1][2], ptsH[0][2], ptsF[0][2],
       ptsH[1][2], crossF, nullptr, Nlev[2], Nlev[2], 128);

  // --- assemble outputs: (out,) + feats1 + feats2 flattened ---
  float* out = (float*)d_out;
  size_t off = 0;
  auto cp = [&](const float* src, size_t n) {
    (void)hipMemcpyAsync(out + off, src, n * sizeof(float), hipMemcpyDeviceToDevice, stream);
    off += n;
  };
  cp(crossF, (size_t)WB * 256 * 128);
  for (int s = 0; s < 2; ++s)
    for (int l = 0; l < 3; ++l) {
      cp(xyzB[s][l], (size_t)WB * Nlev[l] * 3);
      cp(ptsF[s][l], (size_t)WB * Nlev[l] * dlev[l]);
    }
}
